// SelfAttention_44676249813374
// MI455X (gfx1250) — matrix operand order repaired and checked by reading, not compiled, not run on hardware
//
#include <hip/hip_runtime.h>
#include <hip/hip_bf16.h>

typedef __bf16 bf16;
typedef __attribute__((ext_vector_type(16))) bf16 bf16x16;
typedef __attribute__((ext_vector_type(8)))  bf16 bf16x8;
typedef __attribute__((ext_vector_type(8)))  float floatx8;

#define BDIM 4
#define SDIM 2048
#define EDIM 512
#define HDIM 8
#define DDIM 64
#define BS   (BDIM * SDIM)  // 8192

static __device__ __forceinline__ floatx8 wmma_bf(bf16x16 a, bf16x16 b, floatx8 c) {
  // D = A(16x32) * B(32x16) + C, fp32 accumulate
  return __builtin_amdgcn_wmma_f32_16x16x32_bf16(false, a, false, b, (short)0, c, false, false);
}

// A-matrix tile (16 rows x 32 K) from row-major bf16, leading dim `ld` (halfs).
// Lane l: row = row0 + (l&15); halfs 0..15 = 16 contiguous K at k0+(l>>4)*16.
static __device__ __forceinline__ bf16x16 load_a(const bf16* base, int row0, int ld, int k0) {
  int lane = threadIdx.x & 31;
  const bf16* p = base + (size_t)(row0 + (lane & 15)) * ld + k0 + ((lane >> 4) << 4);
  bf16x8 lo = *reinterpret_cast<const bf16x8*>(p);
  bf16x8 hi = *reinterpret_cast<const bf16x8*>(p + 8);
  bf16x16 v;
#pragma unroll
  for (int i = 0; i < 8; ++i) { v[i] = lo[i]; v[8 + i] = hi[i]; }
  return v;
}

// B-matrix tile (32 K x 16 cols). Source stored row-major by COLUMN index (i.e. B^T rows),
// leading dim `ld`. Lane l: col = col0 + (l&15); halfs 0..15 = 16 contiguous K at k0+(l>>4)*16.
static __device__ __forceinline__ bf16x16 load_b(const bf16* baseT, int col0, int ld, int k0) {
  int lane = threadIdx.x & 31;
  const bf16* p = baseT + (size_t)(col0 + (lane & 15)) * ld + k0 + ((lane >> 4) << 4);
  bf16x8 lo = *reinterpret_cast<const bf16x8*>(p);
  bf16x8 hi = *reinterpret_cast<const bf16x8*>(p + 8);
  bf16x16 v;
#pragma unroll
  for (int i = 0; i < 8; ++i) { v[i] = lo[i]; v[8 + i] = hi[i]; }
  return v;
}

__global__ void attn_cvt_x(const float* __restrict__ x, bf16* __restrict__ xb, int n) {
  int i = blockIdx.x * blockDim.x + threadIdx.x;
  if (i < n) xb[i] = (bf16)x[i];
}

// wT[n*512 + k] = w[k*512 + n]  (bf16) : columns become contiguous rows for load_b
__global__ void attn_cvt_wT(const float* __restrict__ w, bf16* __restrict__ wT) {
  int i = blockIdx.x * blockDim.x + threadIdx.x;
  if (i < EDIM * EDIM) {
    int n = i >> 9, k = i & 511;
    wT[i] = (bf16)w[k * EDIM + n];
  }
}

// One wave computes a 32x64 output tile of one projection (z: 0=Q,1=K,2=V).
// Two A-tiles reuse each weight B-tile -> 8 WMMA per 12 b128 loads per K-step.
// Q,K -> [B,H,S,D] bf16 ; V -> transposed [B,H,D,S] bf16.
__global__ void attn_qkv_gemm(const bf16* __restrict__ xb, const bf16* __restrict__ wT,
                              const float* __restrict__ bq, const float* __restrict__ bk,
                              const float* __restrict__ bv,
                              bf16* __restrict__ q, bf16* __restrict__ k, bf16* __restrict__ vt) {
  int wave = threadIdx.x >> 5;
  int lane = threadIdx.x & 31;
  int m0 = (blockIdx.x * 4 + wave) * 32;  // 32-row tile within BS
  int h  = blockIdx.y;
  int z  = blockIdx.z;
  const bf16* w = wT + (size_t)z * EDIM * EDIM;
  const float* bias = (z == 0) ? bq : (z == 1) ? bk : bv;

  floatx8 acc[2][4] = {{{}, {}, {}, {}}, {{}, {}, {}, {}}};
  for (int kk = 0; kk < EDIM; kk += 32) {
    bf16x16 a0 = load_a(xb, m0,      EDIM, kk);
    bf16x16 a1 = load_a(xb, m0 + 16, EDIM, kk);
#pragma unroll
    for (int t = 0; t < 4; ++t) {
      bf16x16 bt = load_b(w, h * DDIM + t * 16, EDIM, kk);
      acc[0][t] = wmma_bf(a0, bt, acc[0][t]);
      acc[1][t] = wmma_bf(a1, bt, acc[1][t]);
    }
  }

  int nl = lane & 15, kh = lane >> 4;
  int bidx = m0 >> 11;                 // 2048 rows per batch; 32-row tile never straddles
  size_t hb = (size_t)bidx * HDIM + h;
#pragma unroll
  for (int u = 0; u < 2; ++u) {
    int s0 = ((m0 + u * 16) & 2047) + kh * 8;
#pragma unroll
    for (int t = 0; t < 4; ++t) {
      int d = t * 16 + nl;
      float bb = bias[h * DDIM + d];
      if (z < 2) {
        bf16* dst = (z == 0) ? q : k;
#pragma unroll
        for (int r = 0; r < 8; ++r)
          dst[(hb * SDIM + s0 + r) * DDIM + d] = (bf16)(acc[u][t][r] + bb);
      } else {
        bf16x8 pk;  // consecutive r -> consecutive s: contiguous 16B store into V^T
#pragma unroll
        for (int r = 0; r < 8; ++r) pk[r] = (bf16)(acc[u][t][r] + bb);
        *reinterpret_cast<bf16x8*>(vt + (hb * DDIM + d) * SDIM + s0) = pk;
      }
    }
  }
}

// Flash attention, computed transposed: S^T = K * Q^T ; ctx^T = V^T * P^T.
// One wave owns 32 queries (two 16-query N-tiles); K and V^T tiles are loaded
// once per key-block and reused by both query tiles: 16 WMMA per 16 b128 loads.
// P^T flows from the scores C-accumulator straight into the next WMMA's B
// operand with a single shfl_xor(16) — no LDS round-trip.
__global__ void attn_flash(const bf16* __restrict__ q, const bf16* __restrict__ k,
                           const bf16* __restrict__ vt, float* __restrict__ out) {
  int wave = threadIdx.x >> 5;
  int lane = threadIdx.x & 31;
  int b = blockIdx.y, h = blockIdx.z;
  int q0 = (blockIdx.x * 4 + wave) * 32;
  size_t bh = (size_t)b * HDIM + h;
  const bf16* qh = q  + bh * SDIM * DDIM;
  const bf16* kh = k  + bh * SDIM * DDIM;
  const bf16* vh = vt + bh * DDIM * SDIM;
  int khf = lane >> 4;

  // Q^T B-operand tiles: [query tile u][d chunk c], resident for the whole loop
  bf16x16 qB[2][2];
#pragma unroll
  for (int u = 0; u < 2; ++u) {
    qB[u][0] = load_b(qh, q0 + u * 16, DDIM, 0);
    qB[u][1] = load_b(qh, q0 + u * 16, DDIM, 32);
  }

  floatx8 acc[2][4] = {{{}, {}, {}, {}}, {{}, {}, {}, {}}};  // ctx^T per query tile
  float m_i[2] = {-3.0e38f, -3.0e38f};
  float l_i[2] = {0.0f, 0.0f};

  for (int j0 = 0; j0 < SDIM; j0 += 32) {
    // K A-tiles for this key block, shared by both query tiles
    bf16x16 kA00 = load_a(kh, j0,      DDIM, 0);
    bf16x16 kA01 = load_a(kh, j0,      DDIM, 32);
    bf16x16 kA10 = load_a(kh, j0 + 16, DDIM, 0);
    bf16x16 kA11 = load_a(kh, j0 + 16, DDIM, 32);

    bf16x16 pB[2];
    float alpha[2];
#pragma unroll
    for (int u = 0; u < 2; ++u) {
      floatx8 s0 = {}, s1 = {};
      s0 = wmma_bf(kA00, qB[u][0], s0);
      s0 = wmma_bf(kA01, qB[u][1], s0);
      s1 = wmma_bf(kA10, qB[u][0], s1);
      s1 = wmma_bf(kA11, qB[u][1], s1);

      float bm = -3.0e38f;
#pragma unroll
      for (int r = 0; r < 8; ++r) {
        s0[r] *= 8.0f; s1[r] *= 8.0f;   // x8 = reference's divide by D^-0.5
        bm = fmaxf(bm, fmaxf(s0[r], s1[r]));
      }
      bm = fmaxf(bm, __shfl_xor(bm, 16, 32));  // lane and lane^16 share a query
      float m_new = fmaxf(m_i[u], bm);
      alpha[u] = __expf(m_i[u] - m_new);

      float p0[8], p1[8], rs = 0.0f;
#pragma unroll
      for (int r = 0; r < 8; ++r) {
        p0[r] = __expf(s0[r] - m_new);
        p1[r] = __expf(s1[r] - m_new);
        rs += p0[r] + p1[r];
      }
      rs += __shfl_xor(rs, 16, 32);
      l_i[u] = l_i[u] * alpha[u] + rs;
      m_i[u] = m_new;

      // Build P^T B-tile: lane needs keys khf*16..khf*16+15; own C-tiles hold
      // keys khf*8+r (s0) and 16+khf*8+r (s1); partner lane^16 holds the rest.
#pragma unroll
      for (int r = 0; r < 8; ++r) {
        float send = khf ? p0[r] : p1[r];
        float recv = __shfl_xor(send, 16, 32);
        pB[u][r]     = (bf16)(khf ? recv : p0[r]);
        pB[u][8 + r] = (bf16)(khf ? p1[r] : recv);
      }
#pragma unroll
      for (int t = 0; t < 4; ++t) {
#pragma unroll
        for (int r = 0; r < 8; ++r) acc[u][t][r] *= alpha[u];
      }
    }

    // ctx^T accumulation: V^T A-tiles loaded once, used by both query tiles
#pragma unroll
    for (int t = 0; t < 4; ++t) {
      bf16x16 vA = load_a(vh, t * 16, SDIM, j0);   // V^T rows: contiguous keys
      acc[0][t] = wmma_bf(vA, pB[0], acc[0][t]);
      acc[1][t] = wmma_bf(vA, pB[1], acc[1][t]);
    }
  }

  // out[b, s, h*64 + d] = ctx^T[d][query] / l ; consecutive r -> consecutive d
#pragma unroll
  for (int u = 0; u < 2; ++u) {
    float inv = 1.0f / l_i[u];
    int s = q0 + u * 16 + (lane & 15);
    size_t obase = ((size_t)b * SDIM + s) * EDIM + h * DDIM + khf * 8;
#pragma unroll
    for (int t = 0; t < 4; ++t) {
      float4 lo = make_float4(acc[u][t][0] * inv, acc[u][t][1] * inv,
                              acc[u][t][2] * inv, acc[u][t][3] * inv);
      float4 hi = make_float4(acc[u][t][4] * inv, acc[u][t][5] * inv,
                              acc[u][t][6] * inv, acc[u][t][7] * inv);
      *reinterpret_cast<float4*>(out + obase + t * 16)     = lo;
      *reinterpret_cast<float4*>(out + obase + t * 16 + 4) = hi;
    }
  }
}

extern "C" void kernel_launch(void* const* d_in, const int* in_sizes, int n_in,
                              void* d_out, int out_size, void* d_ws, size_t ws_size,
                              hipStream_t stream) {
  const float* x  = (const float*)d_in[0];
  const float* wq = (const float*)d_in[1];
  const float* bq = (const float*)d_in[2];
  const float* wk = (const float*)d_in[3];
  const float* bk = (const float*)d_in[4];
  const float* wv = (const float*)d_in[5];
  const float* bv = (const float*)d_in[6];
  float* out = (float*)d_out;

  // workspace layout (bytes)
  char* ws = (char*)d_ws;
  const size_t XB_BYTES = (size_t)BS * EDIM * 2;        // 8 MB
  const size_t WT_BYTES = (size_t)EDIM * EDIM * 2;      // 512 KB each
  const size_t QK_BYTES = (size_t)BS * EDIM * 2;        // 8 MB each
  bf16* xb  = (bf16*)ws;
  bf16* wT  = (bf16*)(ws + XB_BYTES);
  bf16* qb  = (bf16*)(ws + XB_BYTES + 3 * WT_BYTES);
  bf16* kb  = (bf16*)(ws + XB_BYTES + 3 * WT_BYTES + QK_BYTES);
  bf16* vtb = (bf16*)(ws + XB_BYTES + 3 * WT_BYTES + 2 * QK_BYTES);

  attn_cvt_x<<<(BS * EDIM + 255) / 256, 256, 0, stream>>>(x, xb, BS * EDIM);
  attn_cvt_wT<<<(EDIM * EDIM + 255) / 256, 256, 0, stream>>>(wq, wT);
  attn_cvt_wT<<<(EDIM * EDIM + 255) / 256, 256, 0, stream>>>(wk, wT + (size_t)EDIM * EDIM);
  attn_cvt_wT<<<(EDIM * EDIM + 255) / 256, 256, 0, stream>>>(wv, wT + (size_t)2 * EDIM * EDIM);

  attn_qkv_gemm<<<dim3(BS / 128, HDIM, 3), 128, 0, stream>>>(xb, wT, bq, bk, bv, qb, kb, vtb);
  attn_flash<<<dim3(SDIM / 128, BDIM, HDIM), 128, 0, stream>>>(qb, kb, vtb, out);
}
